// STLROneVAE_75857712382250
// MI455X (gfx1250) — compile-verified, matrix operand order checked by reading, not run on hardware
//
#include <hip/hip_runtime.h>

typedef __bf16 bf16;
typedef __attribute__((ext_vector_type(16))) __bf16 v16bf;
typedef __attribute__((ext_vector_type(8)))  __bf16 v8bf;
typedef __attribute__((ext_vector_type(8)))  float  v8f;

#define DEVFN __device__ __forceinline__

static constexpr int  kB  = 256;   // batch
static constexpr int  kT  = 200;   // time
static constexpr int  kH  = 256;   // RNN hidden
static constexpr int  kL  = 128;   // latent
static constexpr long kTB = (long)kB * kT;   // 51200 rows

// ---------- scalar helpers ----------
DEVFN bf16 f2bf(float x) {
  union { float f; unsigned u; } c; c.f = x;
  unsigned r = c.u + 0x7FFFu + ((c.u >> 16) & 1u);   // RNE
  union { unsigned short s; bf16 b; } o; o.s = (unsigned short)(r >> 16);
  return o.b;
}
DEVFN float bf2f(bf16 b) {
  union { bf16 b; unsigned short s; } i; i.b = b;
  union { unsigned u; float f; } o; o.u = (unsigned)i.s << 16;
  return o.f;
}
DEVFN float actf(int act, float x) {
  if (act == 1) return fmaxf(x, 0.f);                       // relu
  if (act == 2) return tanhf(x);                            // tanh
  if (act == 3) return 1.f / (1.f + __expf(-x));            // sigmoid
  if (act == 4) return (x > 20.f) ? x : log1pf(__expf(x));  // softplus
  return x;
}

// fragment = 16 bf16 = two 16-byte vector loads (global_load_b128 / ds_load_b128)
union Frag { v16bf v; v8bf h[2]; };
DEVFN void load_frag(Frag& f, const bf16* __restrict__ row, int kb) {
  f.h[0] = *(const v8bf*)(row + kb);
  f.h[1] = *(const v8bf*)(row + kb + 16);
}

// ---------- generic WMMA GEMM: act(A @ W^T + bias [+ Cin]) ----------
// A: bf16 (M,K); amode=1: logical row r=(t*B+b) sourced from A[(b*T+t)*lda]
// W: bf16 (N,K) row-major. Each wave computes a PAIR of adjacent N-tiles
// sharing one A fragment (2 accumulators, 2 WMMAs per A load).
struct GemmP {
  const bf16* A; long lda; int amode;
  const bf16* W; long ldw;
  const float* bias;
  const bf16* Cin; long ldcin;     // optional bf16 accumulate input
  float* Cout; long ldc; int act;  // optional f32 output
  bf16* Cout2; long ldc2; int act2;// optional bf16 output
  long M, N; int K;
};

__global__ __launch_bounds__(256) void gemm_wmma(GemmP p) {
  const int wave = threadIdx.x >> 5, lane = threadIdx.x & 31;
  const int lg = lane >> 4, li = lane & 15;
  const long pairsN = p.N >> 5;                       // N/32
  const long w = (long)blockIdx.x * 8 + wave;
  if (w >= (p.M >> 4) * pairsN) return;
  const long tm = w / pairsN, tp = w % pairsN;

  const long row = tm * 16 + li;
  const bf16* arow;
  if (p.amode == 1) {                                  // (B,T,dv) -> (T*B, dv)
    long t = row / kB, b = row - t * kB;
    arow = p.A + (b * (long)kT + t) * p.lda;
  } else {
    arow = p.A + row * p.lda;
  }
  const long n0 = tp * 32 + li;
  const bf16* w0 = p.W + n0 * p.ldw;
  const bf16* w1 = p.W + (n0 + 16) * p.ldw;

  v8f acc0 = {}, acc1 = {};
  for (int k0 = 0; k0 < p.K; k0 += 32) {
    const int kb = k0 + lg * 8;
    __builtin_prefetch(w0 + kb + 32, 0, 3);            // near-scope prefetch
    Frag a, b0, b1;
    load_frag(a, arow, kb);
    load_frag(b0, w0, kb);
    load_frag(b1, w1, kb);
    acc0 = __builtin_amdgcn_wmma_f32_16x16x32_bf16(false, a.v, false, b0.v,
                                                   (short)0, acc0, false, false);
    acc1 = __builtin_amdgcn_wmma_f32_16x16x32_bf16(false, a.v, false, b1.v,
                                                   (short)0, acc1, false, false);
  }
#pragma unroll
  for (int j = 0; j < 8; ++j) {
    const long m = tm * 16 + j + lg * 8;
#pragma unroll
    for (int q = 0; q < 2; ++q) {
      const long n = tp * 32 + q * 16 + li;
      float v = (q ? acc1[j] : acc0[j]);
      if (p.bias) v += p.bias[n];
      if (p.Cin)  v += bf2f(p.Cin[m * p.ldcin + n]);
      if (p.Cout)  p.Cout[m * p.ldc + n]   = actf(p.act, v);
      if (p.Cout2) p.Cout2[m * p.ldc2 + n] = f2bf(actf(p.act2, v));
    }
  }
}

// ---------- persistent RNN: h_t = tanh(ih_t + h_{t-1} @ Whh^T + bhh) ----------
// 6 workgroups (view x dir), 16 waves. Recurrent state double-buffered in LDS
// (2 x 128KB) so the step-to-step dependency never leaves the WGP; the global
// bf16 ih buffer is overwritten in place by h for the downstream combiner.
struct RnnP {
  bf16* ih; long seqStride;            // [6][T][B][H] bf16
  const bf16* whhF; const bf16* whhB;
  const float* bhhF; const float* bhhB;
};

__global__ __launch_bounds__(512) void rnn_seq(RnnP p) {
  extern __shared__ char smem[];
  bf16* sH0 = (bf16*)smem;                 // 256x256 bf16 = 128KB
  bf16* sH1 = (bf16*)(smem + 131072);      // 128KB
  const int s = blockIdx.x, dir = s & 1;
  bf16* base = p.ih + (long)s * p.seqStride;
  const bf16*  Whh = dir ? p.whhB : p.whhF;
  const float* bhh = dir ? p.bhhB : p.bhhF;
  const int wave = threadIdx.x >> 5, lane = threadIdx.x & 31;
  const int lg = lane >> 4, li = lane & 15;
  const long stepE = (long)kB * kH;

  int cur = 0;
  for (int step = 0; step < kT; ++step) {
    const int t = dir ? (kT - 1 - step) : step;
    bf16* out = base + (long)t * stepE;
    const bf16* hprev = cur ? sH1 : sH0;
    bf16*       hnext = cur ? sH0 : sH1;
    for (int wk = wave; wk < 128; wk += 16) {          // (B/16) x (H/32) pairs
      const int tm = wk >> 3, tp = wk & 7;
      v8f acc0 = {}, acc1 = {};
      if (step) {
        const bf16* arow = hprev + (tm * 16 + li) * kH;          // LDS
        const bf16* w0 = Whh + (long)(tp * 32 + li) * kH;        // global (WGP$)
        const bf16* w1 = Whh + (long)(tp * 32 + 16 + li) * kH;
        for (int k0 = 0; k0 < kH; k0 += 32) {
          const int kb = k0 + lg * 8;
          __builtin_prefetch(w0 + kb + 32, 0, 3);
          Frag a, b0, b1;
          load_frag(a, arow, kb);
          load_frag(b0, w0, kb);
          load_frag(b1, w1, kb);
          acc0 = __builtin_amdgcn_wmma_f32_16x16x32_bf16(false, a.v, false, b0.v,
                                                         (short)0, acc0, false, false);
          acc1 = __builtin_amdgcn_wmma_f32_16x16x32_bf16(false, a.v, false, b1.v,
                                                         (short)0, acc1, false, false);
        }
      }
#pragma unroll
      for (int j = 0; j < 8; ++j) {
        const long m = tm * 16 + j + lg * 8;
#pragma unroll
        for (int q = 0; q < 2; ++q) {
          const long n = tp * 32 + q * 16 + li;
          float v = (q ? acc1[j] : acc0[j]) + bf2f(out[m * kH + n]) + bhh[n];
          bf16 hv = f2bf(tanhf(v));                    // out held ih[t]
          out[m * kH + n]   = hv;                      // for combiner (later kernel)
          hnext[m * kH + n] = hv;                      // recurrent path stays in LDS
        }
      }
    }
    __syncthreads();                                   // LDS handoff only
    cur ^= 1;
  }
}

// ---------- persistent combiner (scan over t), one workgroup per view ----------
// transW + muW staged in LDS (64KB each); per-step hc intermediate lives in LDS
// (128KB) so phase1 -> phase2 handoff needs only s_barrier, no global fence.
struct CombP {
  const bf16* ih; long seqStride;                  // hf = seq 2v, hb = seq 2v+1
  const bf16* transW; const float* transb;         // (H, L)
  const bf16* muW;    const float* mub;            // (L, H)
  const bf16* sigW;   const float* sigb;           // (L, H)
  const float* eps[3];
  bf16* zbf[3];                                    // bf16 copy of z (T,B,L)
  float* zo[3]; float* muo[3]; float* sdo[3];      // f32 outputs (T,B,L)
};

__global__ __launch_bounds__(512) void combine_seq(CombP p) {
  extern __shared__ char smem[];
  bf16* sTr = (bf16*)smem;                   // 256x128 = 64KB
  bf16* sMu = (bf16*)(smem + 65536);         // 128x256 = 64KB
  bf16* sHc = (bf16*)(smem + 131072);        // 256x256 = 128KB
  const int v = blockIdx.x;
  const bf16* hf = p.ih + (long)(2 * v) * p.seqStride;
  const bf16* hb = p.ih + (long)(2 * v + 1) * p.seqStride;
  const int tid = threadIdx.x;
  for (int i = tid; i < (kH * kL) / 8; i += 512)
    ((v8bf*)sTr)[i] = ((const v8bf*)p.transW)[i];
  for (int i = tid; i < (kL * kH) / 8; i += 512)
    ((v8bf*)sMu)[i] = ((const v8bf*)p.muW)[i];
  __syncthreads();

  const int wave = tid >> 5, lane = tid & 31;
  const int lg = lane >> 4, li = lane & 15;
  const long stepH = (long)kB * kH, stepL = (long)kB * kL;

  for (int t = 0; t < kT; ++t) {
    const bf16* zprev = t ? p.zbf[v] + (long)(t - 1) * stepL : nullptr;
    // phase 1: hc = (tanh(zprev @ transW^T + b) + hf_t + hb_t)/3   (BxH, K=L)
    for (int wk = wave; wk < 128; wk += 16) {          // (B/16) x (H/32) pairs
      const int tm = wk >> 3, tp = wk & 7;
      v8f acc0 = {}, acc1 = {};
      if (zprev) {
        const bf16* arow = zprev + (long)(tm * 16 + li) * kL;    // global
        const bf16* w0 = sTr + (tp * 32 + li) * kL;              // LDS
        const bf16* w1 = sTr + (tp * 32 + 16 + li) * kL;
        for (int k0 = 0; k0 < kL; k0 += 32) {
          const int kb = k0 + lg * 8;
          Frag a, b0, b1;
          load_frag(a, arow, kb);
          load_frag(b0, w0, kb);
          load_frag(b1, w1, kb);
          acc0 = __builtin_amdgcn_wmma_f32_16x16x32_bf16(false, a.v, false, b0.v,
                                                         (short)0, acc0, false, false);
          acc1 = __builtin_amdgcn_wmma_f32_16x16x32_bf16(false, a.v, false, b1.v,
                                                         (short)0, acc1, false, false);
        }
      }
#pragma unroll
      for (int j = 0; j < 8; ++j) {
        const long m = tm * 16 + j + lg * 8;
#pragma unroll
        for (int q = 0; q < 2; ++q) {
          const long n = tp * 32 + q * 16 + li;
          const long idx = (long)t * stepH + m * kH + n;
          float tmp = tanhf((q ? acc1[j] : acc0[j]) + p.transb[n]);
          sHc[m * kH + n] =
              f2bf((tmp + bf2f(hf[idx]) + bf2f(hb[idx])) * (1.f / 3.f));
        }
      }
    }
    __syncthreads();                                   // LDS handoff
    // phase 2: mu/std/z from hc (BxL, K=H); A fragment shared by mu & sig GEMMs
    for (int wk = wave; wk < 128; wk += 16) {          // (B/16) x (L/16)
      const int tm = wk >> 3, tn = wk & 7;
      v8f am = {}, as = {};
      const bf16* arow = sHc + (tm * 16 + li) * kH;              // LDS
      const bf16* wmu = sMu + (tn * 16 + li) * kH;               // LDS
      const bf16* wsg = p.sigW + (long)(tn * 16 + li) * kH;      // global (WGP$)
      for (int k0 = 0; k0 < kH; k0 += 32) {
        const int kb = k0 + lg * 8;
        Frag a, b1, b2;
        load_frag(a, arow, kb);
        load_frag(b1, wmu, kb);
        load_frag(b2, wsg, kb);
        am = __builtin_amdgcn_wmma_f32_16x16x32_bf16(false, a.v, false, b1.v,
                                                     (short)0, am, false, false);
        as = __builtin_amdgcn_wmma_f32_16x16x32_bf16(false, a.v, false, b2.v,
                                                     (short)0, as, false, false);
      }
#pragma unroll
      for (int j = 0; j < 8; ++j) {
        const long m = tm * 16 + j + lg * 8, n = tn * 16 + li;
        const long o = (long)t * stepL + m * kL + n;
        float mu = am[j] + p.mub[n];
        float sd = actf(4, as[j] + p.sigb[n]);
        float z  = mu + sd * p.eps[v][o];
        p.muo[v][o] = mu;
        p.sdo[v][o] = sd;
        p.zo[v][o]  = z;
        p.zbf[v][o] = f2bf(z);
      }
    }
    __threadfence();                                   // z(t) global -> z(t+1) read
    __syncthreads();
  }
}

// ---------- small elementwise kernels ----------
__global__ void f2bf_kernel(const float* __restrict__ s, bf16* __restrict__ d, long n) {
  long i = (long)blockIdx.x * blockDim.x + threadIdx.x;
  if (i < n) d[i] = f2bf(s[i]);
}
__global__ void shift_bf_kernel(const bf16* __restrict__ z, bf16* __restrict__ zp,
                                long total, long headZero) {
  long i = (long)blockIdx.x * blockDim.x + threadIdx.x;
  if (i < total) zp[i] = (i < headZero) ? f2bf(0.f) : z[i - headZero];
}
__global__ void mu_prior_kernel(const float* __restrict__ g, const float* __restrict__ mz,
                                const float* __restrict__ hp, float* __restrict__ o, long n) {
  long i = (long)blockIdx.x * blockDim.x + threadIdx.x;
  if (i < n) { float gg = g[i]; o[i] = (1.f - gg) * mz[i] + gg * hp[i]; }
}

// ---------- host ----------
extern "C" void kernel_launch(void* const* d_in, const int* in_sizes, int n_in,
                              void* d_out, int out_size, void* d_ws, size_t ws_size,
                              hipStream_t stream) {
  (void)in_sizes; (void)n_in; (void)out_size; (void)ws_size;
  const int dvs[3] = {64, 96, 128};
  const float* x[3]   = {(const float*)d_in[0], (const float*)d_in[2], (const float*)d_in[4]};
  const float* eps[3] = {(const float*)d_in[1], (const float*)d_in[3], (const float*)d_in[5]};

  // params flattened: top-level dict order then sorted-key pytree order
  int p = 6;
  const float *f1W[3], *f1b[3], *f2W[3], *f2b[3];
  for (int v = 0; v < 3; ++v) { f1W[v] = (const float*)d_in[p++]; f1b[v] = (const float*)d_in[p++]; }
  for (int v = 0; v < 3; ++v) { f2W[v] = (const float*)d_in[p++]; f2b[v] = (const float*)d_in[p++]; }
  const float* gate1W = (const float*)d_in[p++]; const float* gate1b = (const float*)d_in[p++];
  const float* gate2W = (const float*)d_in[p++]; const float* gate2b = (const float*)d_in[p++];
  const float* muW    = (const float*)d_in[p++]; const float* mub    = (const float*)d_in[p++];
  const float *muE0W[3], *muE0b[3], *muE1W[3], *muE1b[3];
  for (int v = 0; v < 3; ++v) {
    muE0W[v] = (const float*)d_in[p++]; muE0b[v] = (const float*)d_in[p++];
    muE1W[v] = (const float*)d_in[p++]; muE1b[v] = (const float*)d_in[p++];
  }
  const float* muzW  = (const float*)d_in[p++]; const float* muzb  = (const float*)d_in[p++];
  const float* prop1W = (const float*)d_in[p++]; const float* prop1b = (const float*)d_in[p++];
  const float* prop2W = (const float*)d_in[p++]; const float* prop2b = (const float*)d_in[p++];
  const float* WhhB = (const float*)d_in[p++]; const float* WihB = (const float*)d_in[p++];
  const float* bhhB = (const float*)d_in[p++]; const float* bihB = (const float*)d_in[p++];
  const float* WhhF = (const float*)d_in[p++]; const float* WihF = (const float*)d_in[p++];
  const float* bhhF = (const float*)d_in[p++]; const float* bihF = (const float*)d_in[p++];
  const float *sgE0W[3], *sgE0b[3], *sgE1W[3], *sgE1b[3];
  for (int v = 0; v < 3; ++v) {
    sgE0W[v] = (const float*)d_in[p++]; sgE0b[v] = (const float*)d_in[p++];
    sgE1W[v] = (const float*)d_in[p++]; sgE1b[v] = (const float*)d_in[p++];
  }
  const float* sgzW  = (const float*)d_in[p++]; const float* sgzb  = (const float*)d_in[p++];
  const float* sigW  = (const float*)d_in[p++]; const float* sigb  = (const float*)d_in[p++];
  const float* transW = (const float*)d_in[p++]; const float* transb = (const float*)d_in[p++];
  // last leaf is z_init (zeros) — unused

  size_t wsoff = 0;
  auto walloc = [&](size_t bytes) -> void* {
    void* r = (char*)d_ws + wsoff;
    wsoff = (wsoff + bytes + 255) & ~(size_t)255;
    return r;
  };
  auto cvt = [&](const float* src, long n) -> const bf16* {
    bf16* dst = (bf16*)walloc((size_t)n * sizeof(bf16));
    f2bf_kernel<<<(unsigned)((n + 255) / 256), 256, 0, stream>>>(src, dst, n);
    return dst;
  };

  // weights + raw inputs -> bf16
  const bf16 *bf1W[3], *bf2W[3], *bmuE0W[3], *bmuE1W[3], *bsgE0W[3], *bsgE1W[3], *xbf[3];
  for (int v = 0; v < 3; ++v) xbf[v]  = cvt(x[v], kTB * (long)dvs[v]);
  for (int v = 0; v < 3; ++v) bf1W[v] = cvt(f1W[v], 128L * dvs[v]);
  for (int v = 0; v < 3; ++v) bf2W[v] = cvt(f2W[v], 128L * (288 - dvs[v]));
  const bf16* bWihF = cvt(WihF, 256L * 256); const bf16* bWihB = cvt(WihB, 256L * 256);
  const bf16* bWhhF = cvt(WhhF, 256L * 256); const bf16* bWhhB = cvt(WhhB, 256L * 256);
  const bf16* btransW = cvt(transW, 256L * 128);
  const bf16* bmuW = cvt(muW, 128L * 256); const bf16* bsigW = cvt(sigW, 128L * 256);
  const bf16* bgate1W = cvt(gate1W, 256L * 128); const bf16* bgate2W = cvt(gate2W, 128L * 256);
  const bf16* bprop1W = cvt(prop1W, 256L * 128); const bf16* bprop2W = cvt(prop2W, 128L * 256);
  const bf16* bmuzW = cvt(muzW, 128L * 128); const bf16* bsgzW = cvt(sgzW, 128L * 128);
  for (int v = 0; v < 3; ++v) {
    bmuE0W[v] = cvt(muE0W[v], 256L * 128); bmuE1W[v] = cvt(muE1W[v], (long)dvs[v] * 256);
    bsgE0W[v] = cvt(sgE0W[v], 256L * 128); bsgE1W[v] = cvt(sgE1W[v], (long)dvs[v] * 256);
  }

  // bf16 activation scratch
  bf16* inpbf = (bf16*)walloc(kTB * 256 * 2);       // (T*B, 256) reused per view
  bf16* ihbf  = (bf16*)walloc(6 * kTB * 256 * 2);   // [6][T][B][H], becomes h
  bf16* zbf   = (bf16*)walloc(3L * kTB * 128 * 2);  // bf16 copy of z per view
  bf16* zpbf  = (bf16*)walloc(kTB * 128 * 2);
  bf16* s1bf  = (bf16*)walloc(kTB * 256 * 2);
  bf16* hpbf  = (bf16*)walloc(kTB * 128 * 2);       // relu(hprop) bf16
  // f32 scratch (elementwise consumers)
  float* gbuf = (float*)walloc(kTB * 128 * 4);
  float* hp   = (float*)walloc(kTB * 128 * 4);
  float* mz   = (float*)walloc(kTB * 128 * 4);

  // output layout: per view [z, mu_post, std_post, mu_prior, std_prior, mu_x, std_x]
  float* outp = (float*)d_out;
  float *Zo[3], *MUp[3], *SDp[3], *MUpr[3], *SDpr[3], *MUx[3], *SDx[3];
  long off = 0; const long s128 = kTB * 128;
  for (int v = 0; v < 3; ++v) {
    Zo[v] = outp + off;  off += s128;
    MUp[v] = outp + off; off += s128;
    SDp[v] = outp + off; off += s128;
    MUpr[v] = outp + off; off += s128;
    SDpr[v] = outp + off; off += s128;
    MUx[v] = outp + off; off += kTB * dvs[v];
    SDx[v] = outp + off; off += kTB * dvs[v];
  }

  auto gemm = [&](const bf16* A, long lda, int amode,
                  const bf16* W, long ldw, const float* bias,
                  const bf16* Cin, long ldcin,
                  float* Cout, long ldc, int act,
                  bf16* Cout2, long ldc2, int act2,
                  long M, long N, int K) {
    GemmP g; g.A = A; g.lda = lda; g.amode = amode;
    g.W = W; g.ldw = ldw; g.bias = bias; g.Cin = Cin; g.ldcin = ldcin;
    g.Cout = Cout; g.ldc = ldc; g.act = act;
    g.Cout2 = Cout2; g.ldc2 = ldc2; g.act2 = act2;
    g.M = M; g.N = N; g.K = K;
    unsigned blocks = (unsigned)(((M / 16) * (N / 32) + 7) / 8);
    gemm_wmma<<<blocks, 256, 0, stream>>>(g);
  };

  // ---- input projections + ih precompute (parallel WMMA GEMMs, bf16 out) ----
  for (int v = 0; v < 3; ++v) {
    gemm(xbf[v], dvs[v], 1, bf1W[v], dvs[v], f1b[v], nullptr, 0,
         nullptr, 0, 0, inpbf, 256, 0, kTB, 128, dvs[v]);          // inp[:, :128]
    const int o1 = (v == 0) ? 1 : 0;
    const int o2 = (v == 2) ? 1 : 2;
    const long ldf2 = 288 - dvs[v];
    gemm(xbf[o1], dvs[o1], 1, bf2W[v], ldf2, f2b[v], nullptr, 0,
         nullptr, 0, 0, inpbf + 128, 256, 0, kTB, 128, dvs[o1]);   // X_rest part 1
    gemm(xbf[o2], dvs[o2], 1, bf2W[v] + dvs[o1], ldf2, nullptr, inpbf + 128, 256,
         nullptr, 0, 0, inpbf + 128, 256, 0, kTB, 128, dvs[o2]);   // part 2 (acc)
    bf16* ihF = ihbf + (long)(2 * v) * kTB * 256;
    bf16* ihB = ihbf + (long)(2 * v + 1) * kTB * 256;
    gemm(inpbf, 256, 0, bWihF, 256, bihF, nullptr, 0,
         nullptr, 0, 0, ihF, 256, 0, kTB, 256, 256);
    gemm(inpbf, 256, 0, bWihB, 256, bihB, nullptr, 0,
         nullptr, 0, 0, ihB, 256, 0, kTB, 256, 256);
  }

  // ---- 6 concurrent persistent RNN sequences (LDS-resident recurrence) ----
  (void)hipFuncSetAttribute((const void*)rnn_seq,
                            hipFuncAttributeMaxDynamicSharedMemorySize, 262144);
  RnnP r; r.ih = ihbf; r.seqStride = kTB * 256;
  r.whhF = bWhhF; r.whhB = bWhhB; r.bhhF = bhhF; r.bhhB = bhhB;
  rnn_seq<<<6, 512, 262144, stream>>>(r);

  // ---- 3 concurrent latent scans (weights + hc staged in LDS) ----
  (void)hipFuncSetAttribute((const void*)combine_seq,
                            hipFuncAttributeMaxDynamicSharedMemorySize, 262144);
  CombP c; c.ih = ihbf; c.seqStride = kTB * 256;
  c.transW = btransW; c.transb = transb;
  c.muW = bmuW; c.mub = mub; c.sigW = bsigW; c.sigb = sigb;
  for (int v = 0; v < 3; ++v) {
    c.eps[v] = eps[v];
    c.zbf[v] = zbf + (long)v * kTB * 128;
    c.zo[v] = Zo[v]; c.muo[v] = MUp[v]; c.sdo[v] = SDp[v];
  }
  combine_seq<<<3, 512, 262144, stream>>>(c);

  // ---- DMM prior + emissions (parallel WMMA GEMMs) ----
  const long nzp = kTB * 128;
  for (int v = 0; v < 3; ++v) {
    const bf16* zv = zbf + (long)v * kTB * 128;
    shift_bf_kernel<<<(unsigned)((nzp + 255) / 256), 256, 0, stream>>>(
        zv, zpbf, nzp, (long)kB * 128);
    gemm(zpbf, 128, 0, bgate1W, 128, gate1b, nullptr, 0,
         nullptr, 0, 0, s1bf, 256, 1, kTB, 256, 128);              // relu -> bf16
    gemm(s1bf, 256, 0, bgate2W, 256, gate2b, nullptr, 0,
         gbuf, 128, 3, nullptr, 0, 0, kTB, 128, 256);              // sigmoid f32
    gemm(zpbf, 128, 0, bprop1W, 128, prop1b, nullptr, 0,
         nullptr, 0, 0, s1bf, 256, 1, kTB, 256, 128);              // relu -> bf16
    gemm(s1bf, 256, 0, bprop2W, 256, prop2b, nullptr, 0,
         hp, 128, 0, hpbf, 128, 1, kTB, 128, 256);                 // hp f32 + relu bf16
    gemm(zpbf, 128, 0, bmuzW, 128, muzb, nullptr, 0,
         mz, 128, 0, nullptr, 0, 0, kTB, 128, 128);                // muz f32
    mu_prior_kernel<<<(unsigned)((nzp + 255) / 256), 256, 0, stream>>>(
        gbuf, mz, hp, MUpr[v], nzp);
    gemm(hpbf, 128, 0, bsgzW, 128, sgzb, nullptr, 0,
         SDpr[v], 128, 4, nullptr, 0, 0, kTB, 128, 128);           // softplus f32
    // emission nets
    gemm(zv, 128, 0, bmuE0W[v], 128, muE0b[v], nullptr, 0,
         nullptr, 0, 0, s1bf, 256, 1, kTB, 256, 128);
    gemm(s1bf, 256, 0, bmuE1W[v], 256, muE1b[v], nullptr, 0,
         MUx[v], dvs[v], 0, nullptr, 0, 0, kTB, dvs[v], 256);
    gemm(zv, 128, 0, bsgE0W[v], 128, sgE0b[v], nullptr, 0,
         nullptr, 0, 0, s1bf, 256, 1, kTB, 256, 128);
    gemm(s1bf, 256, 0, bsgE1W[v], 256, sgE1b[v], nullptr, 0,
         SDx[v], dvs[v], 4, nullptr, 0, 0, kTB, dvs[v], 256);
  }
}
